// CrossAttnFusionNet_38843684225565
// MI455X (gfx1250) — matrix operand order repaired and checked by reading, not compiled, not run on hardware
//
#include <hip/hip_runtime.h>
#include <hip/hip_bf16.h>

// ---------------------------------------------------------------------------
// CrossAttnFusionNet for MI455X (gfx1250), wave32 + v_wmma_f32_16x16x32_bf16.
// seq_len==1 MHA collapses (softmax over one key == 1), so each branch is
// folded into a single 128x128 matrix Weff = Wo @ Wv (+ beff).
// Pipeline: prep (f32->bf16 weight convert + fold)  ->
//   k_proj x3 (GEMM+LN+ReLU -> tva bf16) -> k_ctx (6 folded GEMMs -> feats)
//   -> k_gate (VALU gating softmax, in-place scale) -> k_mlp (3 chained GEMMs
//   in LDS -> d_out f32 [B,2]).
// ---------------------------------------------------------------------------

typedef __bf16 bf16_t;
typedef __attribute__((ext_vector_type(16))) __bf16 v16bf;
typedef __attribute__((ext_vector_type(8)))  __bf16 v8bf;
typedef __attribute__((ext_vector_type(8)))  float   v8f;

#define BATCH 32768

// ---- workspace layout (bytes) ---------------------------------------------
static const size_t OFF_WT    = 0;                                   // 128x768 bf16
static const size_t OFF_WV    = OFF_WT   + (size_t)128*768*2;        // 128x512 bf16
static const size_t OFF_WA    = OFF_WV   + (size_t)128*512*2;        // 128x384 bf16
static const size_t OFF_WEFF  = OFF_WA   + (size_t)128*384*2;        // 6x128x128 bf16
static const size_t OFF_BEFF  = OFF_WEFF + (size_t)6*128*128*2;      // 6x128 f32
static const size_t OFF_WF1   = OFF_BEFF + (size_t)6*128*4;          // 256x768 bf16
static const size_t OFF_WF2   = OFF_WF1  + (size_t)256*768*2;        // 256x256 bf16
static const size_t OFF_WOUT  = OFF_WF2  + (size_t)256*256*2;        // 16x256 bf16 (padded)
static const size_t OFF_TVA   = OFF_WOUT + (size_t)16*256*2;         // 3xBx128 bf16
static const size_t OFF_FEATS = OFF_TVA  + (size_t)3*BATCH*128*2;    // Bx768 bf16
static const size_t WS_NEED   = OFF_FEATS+ (size_t)BATCH*768*2;      // ~77 MB

// ---- helpers ---------------------------------------------------------------
__device__ __forceinline__ v8f vzero8f() {
  v8f z;
#pragma unroll
  for (int i = 0; i < 8; ++i) z[i] = 0.0f;
  return z;
}

__device__ __forceinline__ v16bf make_frag(v8bf lo, v8bf hi) {
  v16bf r;
#pragma unroll
  for (int i = 0; i < 8; ++i) { r[i] = lo[i]; r[i + 8] = hi[i]; }
  return r;
}

// A fragment (16x32 bf16 per ISA layout): lane<16 holds K {0..7,16..23},
// lane>=16 holds K {8..15,24..31} of row (lane&15).
__device__ __forceinline__ v16bf load_a_bf16(const bf16_t* base, int ld, int row,
                                             int kb, int half) {
  const bf16_t* p = base + (size_t)row * ld + kb + half * 8;
  v8bf lo = *(const v8bf*)(p);
  v8bf hi = *(const v8bf*)(p + 16);
  return make_frag(lo, hi);
}

// B fragment (32x16 = W.T tile from row-major W[n][k]): lane = column n,
// lanes<16 hold K {0..7,16..23}, lanes>=16 hold K {8..15,24..31} (matches A).
__device__ __forceinline__ v16bf load_b_bf16(const bf16_t* base, int ld, int nrow,
                                             int kb, int half) {
  const bf16_t* p = base + (size_t)nrow * ld + kb + half * 8;
  v8bf lo = *(const v8bf*)(p);
  v8bf hi = *(const v8bf*)(p + 16);
  return make_frag(lo, hi);
}

// A fragment built from f32 global memory with on-the-fly bf16 convert.
__device__ __forceinline__ v16bf load_a_f32cvt(const float* p) {
  float4 a0 = *(const float4*)(p);
  float4 a1 = *(const float4*)(p + 4);
  float4 a2 = *(const float4*)(p + 16);
  float4 a3 = *(const float4*)(p + 20);
  v16bf r;
  r[0]=(bf16_t)a0.x;  r[1]=(bf16_t)a0.y;  r[2]=(bf16_t)a0.z;  r[3]=(bf16_t)a0.w;
  r[4]=(bf16_t)a1.x;  r[5]=(bf16_t)a1.y;  r[6]=(bf16_t)a1.z;  r[7]=(bf16_t)a1.w;
  r[8]=(bf16_t)a2.x;  r[9]=(bf16_t)a2.y;  r[10]=(bf16_t)a2.z; r[11]=(bf16_t)a2.w;
  r[12]=(bf16_t)a3.x; r[13]=(bf16_t)a3.y; r[14]=(bf16_t)a3.z; r[15]=(bf16_t)a3.w;
  return r;
}

__device__ __forceinline__ v8f wmma_bf16(v16bf a, v16bf b, v8f c) {
  return __builtin_amdgcn_wmma_f32_16x16x32_bf16(false, a, false, b, (short)0, c,
                                                 false, false);
}

// ---- prep kernels ----------------------------------------------------------
__global__ void k_cvt(const float* __restrict__ src, bf16_t* __restrict__ dst, int n) {
  int i = blockIdx.x * blockDim.x + threadIdx.x;
  if (i < n) dst[i] = (bf16_t)src[i];
}

__global__ void k_cvt_pad_rows(const float* __restrict__ src, bf16_t* __restrict__ dst,
                               int rows_valid, int rows_total, int cols) {
  int i = blockIdx.x * blockDim.x + threadIdx.x;
  int n = rows_total * cols;
  if (i < n) {
    int r = i / cols;
    dst[i] = (r < rows_valid) ? (bf16_t)src[i] : (bf16_t)0.0f;
  }
}

// Weff[b] = Wo[b] @ Wv[b] (Wv = rows 256..383 of W_qkv[b]); beff = Wo@bv + bo.
__global__ void k_fold(const float* __restrict__ Wqkv, const float* __restrict__ bqkv,
                       const float* __restrict__ Wo, const float* __restrict__ bo,
                       bf16_t* __restrict__ Weff, float* __restrict__ beff) {
  int b = blockIdx.x >> 7;   // 6*128 blocks
  int m = blockIdx.x & 127;
  int k = threadIdx.x;       // 128 threads
  const float* Wv  = Wqkv + ((size_t)b * 384 + 256) * 128;
  const float* Wom = Wo + ((size_t)b * 128 + m) * 128;
  float s = 0.0f;
  for (int c = 0; c < 128; ++c) s += Wom[c] * Wv[(size_t)c * 128 + k];
  Weff[((size_t)b * 128 + m) * 128 + k] = (bf16_t)s;
  if (k == 0) {
    const float* bv = bqkv + b * 384 + 256;
    float t = bo[(size_t)b * 128 + m];
    for (int c = 0; c < 128; ++c) t += Wom[c] * bv[c];
    beff[b * 128 + m] = t;
  }
}

// ---- stage 1: Linear + LayerNorm + ReLU -> tva bf16 [B][128] ---------------
__global__ __launch_bounds__(256)
void k_proj(const float* __restrict__ X, int D, const bf16_t* __restrict__ W,
            const float* __restrict__ bias, const float* __restrict__ lnw,
            const float* __restrict__ lnb, bf16_t* __restrict__ outT) {
  const int wave = threadIdx.x >> 5, lane = threadIdx.x & 31;
  const int half = lane >> 4, n = lane & 15;
  const int rbase = blockIdx.x * 128 + wave * 16;

  v8f acc[8];
#pragma unroll
  for (int j = 0; j < 8; ++j) acc[j] = vzero8f();

  const float* xrow = X + (size_t)(rbase + n) * D;
  for (int kb = 0; kb < D; kb += 32) {
    __builtin_prefetch(xrow + kb + 64, 0, 0);         // global_prefetch_b8
    v16bf af = load_a_f32cvt(xrow + kb + half * 8);
#pragma unroll
    for (int j = 0; j < 8; ++j) {
      v16bf bm = load_b_bf16(W, D, j * 16 + n, kb, half);
      acc[j] = wmma_bf16(af, bm, acc[j]);
    }
  }

  // bias + row stats (rows of a 16x16 C tile live across the 16-lane group)
  float s[8], q[8];
#pragma unroll
  for (int r = 0; r < 8; ++r) { s[r] = 0.0f; q[r] = 0.0f; }
#pragma unroll
  for (int j = 0; j < 8; ++j) {
    float bj = bias[j * 16 + n];
#pragma unroll
    for (int r = 0; r < 8; ++r) {
      float v = acc[j][r] + bj;
      acc[j][r] = v;
      s[r] += v;
      q[r] += v * v;
    }
  }
#pragma unroll
  for (int r = 0; r < 8; ++r) {
    float sv = s[r], qv = q[r];
    sv += __shfl_xor(sv, 1);  qv += __shfl_xor(qv, 1);
    sv += __shfl_xor(sv, 2);  qv += __shfl_xor(qv, 2);
    sv += __shfl_xor(sv, 4);  qv += __shfl_xor(qv, 4);
    sv += __shfl_xor(sv, 8);  qv += __shfl_xor(qv, 8);
    float mean = sv * (1.0f / 128.0f);
    float var  = qv * (1.0f / 128.0f) - mean * mean;
    s[r] = mean;
    q[r] = rsqrtf(var + 1e-5f);
  }
#pragma unroll
  for (int j = 0; j < 8; ++j) {
    float lw = lnw[j * 16 + n], lb = lnb[j * 16 + n];
#pragma unroll
    for (int r = 0; r < 8; ++r) {
      float v = (acc[j][r] - s[r]) * q[r] * lw + lb;
      v = fmaxf(v, 0.0f);
      outT[(size_t)(rbase + half * 8 + r) * 128 + j * 16 + n] = (bf16_t)v;
    }
  }
}

// ---- stage 2: 6 folded branch GEMMs -> feats bf16 [B][768] -----------------
__global__ __launch_bounds__(256)
void k_ctx(const bf16_t* __restrict__ tva, const bf16_t* __restrict__ Weff,
           const float* __restrict__ beff, bf16_t* __restrict__ feats) {
  const int wave = threadIdx.x >> 5, lane = threadIdx.x & 31;
  const int half = lane >> 4, n = lane & 15;
  const int rbase = blockIdx.x * 128 + wave * 16;

  for (int b = 0; b < 6; ++b) {
    int sm = (0x102021 >> (b * 4)) & 0x7;   // source modality {1,2,0,2,0,1}
    const bf16_t* S  = tva + (size_t)sm * BATCH * 128;
    const bf16_t* Wb = Weff + (size_t)b * 128 * 128;
    v8f acc[8];
#pragma unroll
    for (int j = 0; j < 8; ++j) acc[j] = vzero8f();
    for (int kb = 0; kb < 128; kb += 32) {
      v16bf af = load_a_bf16(S, 128, rbase + n, kb, half);
#pragma unroll
      for (int j = 0; j < 8; ++j) {
        v16bf bm = load_b_bf16(Wb, 128, j * 16 + n, kb, half);
        acc[j] = wmma_bf16(af, bm, acc[j]);
      }
    }
#pragma unroll
    for (int j = 0; j < 8; ++j) {
      float bj = beff[b * 128 + j * 16 + n];
#pragma unroll
      for (int r = 0; r < 8; ++r) {
        feats[(size_t)(rbase + half * 8 + r) * 768 + b * 128 + j * 16 + n] =
            (bf16_t)(acc[j][r] + bj);
      }
    }
  }
}

// ---- stage 3: gating softmax + in-place scale (one wave per row) -----------
__global__ __launch_bounds__(256)
void k_gate(bf16_t* __restrict__ feats, const float* __restrict__ Wg,
            const float* __restrict__ bg) {
  __shared__ float wg[6 * 768];
  for (int i = threadIdx.x; i < 6 * 768; i += 256) wg[i] = Wg[i];
  __syncthreads();

  const int wave = threadIdx.x >> 5, lane = threadIdx.x & 31;
  const int row = blockIdx.x * 8 + wave;
  bf16_t* fr = feats + (size_t)row * 768;

  float f[24];
#pragma unroll
  for (int t = 0; t < 24; ++t) f[t] = (float)fr[lane + t * 32];

  float sc[6];
#pragma unroll
  for (int i = 0; i < 6; ++i) {
    float p = 0.0f;
#pragma unroll
    for (int t = 0; t < 24; ++t) p += f[t] * wg[i * 768 + lane + t * 32];
    p += __shfl_xor(p, 1);
    p += __shfl_xor(p, 2);
    p += __shfl_xor(p, 4);
    p += __shfl_xor(p, 8);
    p += __shfl_xor(p, 16);
    sc[i] = p + bg[i];
  }
  float mx = sc[0];
#pragma unroll
  for (int i = 1; i < 6; ++i) mx = fmaxf(mx, sc[i]);
  float se = 0.0f;
#pragma unroll
  for (int i = 0; i < 6; ++i) { sc[i] = __expf(sc[i] - mx); se += sc[i]; }
  float inv = 1.0f / se;
#pragma unroll
  for (int i = 0; i < 6; ++i) sc[i] *= inv;

#pragma unroll
  for (int t = 0; t < 24; ++t) {
    int c = lane + t * 32;
    fr[c] = (bf16_t)(f[t] * sc[c >> 7]);
  }
}

// ---- stage 4: MLP (768->256 relu, 256->256 relu, 256->2) -------------------
__global__ __launch_bounds__(256)
void k_mlp(const bf16_t* __restrict__ gated, const bf16_t* __restrict__ W1,
           const float* __restrict__ b1, const bf16_t* __restrict__ W2,
           const float* __restrict__ b2, const bf16_t* __restrict__ W3,
           const float* __restrict__ b3, float* __restrict__ out) {
  extern __shared__ bf16_t hbuf[];           // [128][264] bf16, padded stride
  const int LDH = 264;
  const int wave = threadIdx.x >> 5, lane = threadIdx.x & 31;
  const int half = lane >> 4, n = lane & 15;
  const int rbase = blockIdx.x * 128 + wave * 16;

  // h1 = relu(gated @ W1.T + b1)  : K=768, N=256 (16 tiles)
  v8f acc[16];
#pragma unroll
  for (int j = 0; j < 16; ++j) acc[j] = vzero8f();
  for (int kb = 0; kb < 768; kb += 32) {
    v16bf af = load_a_bf16(gated, 768, rbase + n, kb, half);
#pragma unroll
    for (int j = 0; j < 16; ++j) {
      v16bf bm = load_b_bf16(W1, 768, j * 16 + n, kb, half);
      acc[j] = wmma_bf16(af, bm, acc[j]);
    }
  }
#pragma unroll
  for (int j = 0; j < 16; ++j) {
    float bj = b1[j * 16 + n];
#pragma unroll
    for (int r = 0; r < 8; ++r) {
      hbuf[(size_t)(wave * 16 + half * 8 + r) * LDH + j * 16 + n] =
          (bf16_t)fmaxf(acc[j][r] + bj, 0.0f);
    }
  }
  __syncthreads();

  // h2 = relu(h1 @ W2.T + b2) : K=256, N=256
#pragma unroll
  for (int j = 0; j < 16; ++j) acc[j] = vzero8f();
  for (int kb = 0; kb < 256; kb += 32) {
    v16bf af = load_a_bf16(hbuf, LDH, wave * 16 + n, kb, half);
#pragma unroll
    for (int j = 0; j < 16; ++j) {
      v16bf bm = load_b_bf16(W2, 256, j * 16 + n, kb, half);
      acc[j] = wmma_bf16(af, bm, acc[j]);
    }
  }
  __syncthreads();   // everyone done reading h1 before overwrite
#pragma unroll
  for (int j = 0; j < 16; ++j) {
    float bj = b2[j * 16 + n];
#pragma unroll
    for (int r = 0; r < 8; ++r) {
      hbuf[(size_t)(wave * 16 + half * 8 + r) * LDH + j * 16 + n] =
          (bf16_t)fmaxf(acc[j][r] + bj, 0.0f);
    }
  }
  __syncthreads();

  // out = h2 @ Wout.T + bout : K=256, N padded to 16 (cols 0..1 valid)
  v8f o = vzero8f();
  for (int kb = 0; kb < 256; kb += 32) {
    v16bf af = load_a_bf16(hbuf, LDH, wave * 16 + n, kb, half);
    v16bf bm = load_b_bf16(W3, 256, n, kb, half);
    o = wmma_bf16(af, bm, o);
  }
  if (n < 2) {
    float bj = b3[n];
#pragma unroll
    for (int r = 0; r < 8; ++r) {
      out[(size_t)(rbase + half * 8 + r) * 2 + n] = o[r] + bj;
    }
  }
}

// ---------------------------------------------------------------------------
extern "C" void kernel_launch(void* const* d_in, const int* in_sizes, int n_in,
                              void* d_out, int out_size, void* d_ws, size_t ws_size,
                              hipStream_t stream) {
  (void)in_sizes; (void)n_in; (void)out_size;
  if (ws_size < WS_NEED) return;   // needs ~77 MB of workspace

  const float* xt    = (const float*)d_in[0];
  const float* xv    = (const float*)d_in[1];
  const float* xa    = (const float*)d_in[2];
  const float* Wt    = (const float*)d_in[3];
  const float* bt    = (const float*)d_in[4];
  const float* lnw_t = (const float*)d_in[5];
  const float* lnb_t = (const float*)d_in[6];
  const float* Wv_   = (const float*)d_in[7];
  const float* bv_   = (const float*)d_in[8];
  const float* lnw_v = (const float*)d_in[9];
  const float* lnb_v = (const float*)d_in[10];
  const float* Wa    = (const float*)d_in[11];
  const float* ba    = (const float*)d_in[12];
  const float* lnw_a = (const float*)d_in[13];
  const float* lnb_a = (const float*)d_in[14];
  const float* Wqkv  = (const float*)d_in[15];
  const float* bqkv  = (const float*)d_in[16];
  const float* Wo    = (const float*)d_in[17];
  const float* bo    = (const float*)d_in[18];
  const float* Wg    = (const float*)d_in[19];
  const float* bg    = (const float*)d_in[20];
  const float* Wf1   = (const float*)d_in[21];
  const float* bf1   = (const float*)d_in[22];
  const float* Wf2   = (const float*)d_in[23];
  const float* bf2   = (const float*)d_in[24];
  const float* Wout  = (const float*)d_in[25];
  const float* bout  = (const float*)d_in[26];

  char* ws = (char*)d_ws;
  bf16_t* Wt_bf   = (bf16_t*)(ws + OFF_WT);
  bf16_t* Wv_bf   = (bf16_t*)(ws + OFF_WV);
  bf16_t* Wa_bf   = (bf16_t*)(ws + OFF_WA);
  bf16_t* Weff_bf = (bf16_t*)(ws + OFF_WEFF);
  float*  beff    = (float*)(ws + OFF_BEFF);
  bf16_t* Wf1_bf  = (bf16_t*)(ws + OFF_WF1);
  bf16_t* Wf2_bf  = (bf16_t*)(ws + OFF_WF2);
  bf16_t* Wout_bf = (bf16_t*)(ws + OFF_WOUT);
  bf16_t* tva     = (bf16_t*)(ws + OFF_TVA);
  bf16_t* feats   = (bf16_t*)(ws + OFF_FEATS);

  // ---- prep: weight conversion + branch folding ----
  k_cvt<<<(128 * 768 + 255) / 256, 256, 0, stream>>>(Wt,  Wt_bf,  128 * 768);
  k_cvt<<<(128 * 512 + 255) / 256, 256, 0, stream>>>(Wv_, Wv_bf,  128 * 512);
  k_cvt<<<(128 * 384 + 255) / 256, 256, 0, stream>>>(Wa,  Wa_bf,  128 * 384);
  k_cvt<<<(256 * 768 + 255) / 256, 256, 0, stream>>>(Wf1, Wf1_bf, 256 * 768);
  k_cvt<<<(256 * 256 + 255) / 256, 256, 0, stream>>>(Wf2, Wf2_bf, 256 * 256);
  k_cvt_pad_rows<<<(16 * 256 + 255) / 256, 256, 0, stream>>>(Wout, Wout_bf, 2, 16, 256);
  k_fold<<<6 * 128, 128, 0, stream>>>(Wqkv, bqkv, Wo, bo, Weff_bf, beff);

  // ---- stage 1: three projection blocks ----
  k_proj<<<BATCH / 128, 256, 0, stream>>>(xt, 768, Wt_bf, bt, lnw_t, lnb_t,
                                          tva + (size_t)0 * BATCH * 128);
  k_proj<<<BATCH / 128, 256, 0, stream>>>(xv, 512, Wv_bf, bv_, lnw_v, lnb_v,
                                          tva + (size_t)1 * BATCH * 128);
  k_proj<<<BATCH / 128, 256, 0, stream>>>(xa, 384, Wa_bf, ba, lnw_a, lnb_a,
                                          tva + (size_t)2 * BATCH * 128);

  // ---- stage 2: 6 folded attention branches -> feats ----
  k_ctx<<<BATCH / 128, 256, 0, stream>>>(tva, Weff_bf, beff, feats);

  // ---- stage 3: gating softmax + scale (in place) ----
  k_gate<<<BATCH / 8, 256, 0, stream>>>(feats, Wg, bg);

  // ---- stage 4: MLP -> d_out ----
  k_mlp<<<BATCH / 128, 256, (size_t)128 * 264 * 2, stream>>>(
      feats, Wf1_bf, bf1, Wf2_bf, bf2, Wout_bf, bout, (float*)d_out);
}
